// DeiT_17351667875874
// MI455X (gfx1250) — compile-verified, matrix operand order checked by reading, not run on hardware
//
#include <hip/hip_runtime.h>
#include <hip/hip_bf16.h>
#include <math.h>

// ---------------------------------------------------------------------------
// DeiT forward for gfx1250 (CDNA5): all GEMMs via v_wmma_f32_16x16x32_f16.
// GEMM core: 8 wave32s/block, 32x64 output per wave (2x4 subtiles, 8 WMMA
// per K-step). K-loop unrolled x2 with ping-pong fragment register sets:
// no rotation moves, loads of step s+1 issue before the WMMAs of step s.
// ---------------------------------------------------------------------------

typedef __attribute__((ext_vector_type(16))) _Float16 v16h;
typedef __attribute__((ext_vector_type(8)))  _Float16 v8h;
typedef __attribute__((ext_vector_type(8)))  float    v8f;

#define BATCH   32
#define CIN     3
#define IMGS    224
#define PSZ     16
#define EMB     768
#define DEPTHL  12
#define NHEAD   12
#define FFD     3072
#define NCLS    4
#define HPD     14
#define NPATCH  196
#define SEQ     198
#define HDIM    64
#define SPAD    224          // SEQ padded to multiple of 32 for WMMA K-loop
#define LN_EPS  1e-5f

enum { EPI_BIAS = 1, EPI_GELU = 2, EPI_RESID = 4, EPI_OUT_F16 = 8, EPI_SCALE = 16 };

// Load one 16x32 f16 WMMA fragment from a row-major [rows, ld] matrix.
// CDNA5 16-bit A/B layout: lanes 0-15 hold K = +0..7 and +16..23,
// lanes 16-31 hold K = +8..15 and +24..31 (the +8 is folded into the pointer).
__device__ __forceinline__ v16h ldfrag(const _Float16* __restrict__ p) {
    v16h f;
    *((v8h*)&f)     = *(const v8h*)p;
    *((v8h*)&f + 1) = *(const v8h*)(p + 16);
    return f;
}

__device__ __forceinline__ v8f wmma_f16(v16h a, v16h b, v8f c) {
    return __builtin_amdgcn_wmma_f32_16x16x32_f16(false, a, false, b, (short)0, c,
                                                  false, false);
}

// C[M,N] = A[M,K] * Bt[N,K]^T (+ epilogue).  NT = active 16-col subtiles.
template <int NT>
__global__ __launch_bounds__(256)
void k_gemm_wmma(const _Float16* __restrict__ A, const _Float16* __restrict__ Bt,
                 void* __restrict__ Cout, const float* __restrict__ resid,
                 const float* __restrict__ bias,
                 int M, int N, int K, int lda, int ldb, int ldc, int nbase,
                 long long sAb, long long sAh, long long sBb, long long sBh,
                 long long sCb, long long sCh, long long sRb,
                 int nhdiv, float scale, int flags) {
    const int z  = blockIdx.z;
    const int bb = z / nhdiv;
    const int hh = z % nhdiv;
    A  += (size_t)bb * sAb + (size_t)hh * sAh;
    Bt += (size_t)bb * sBb + (size_t)hh * sBh;
    float*    Cf = (float*)Cout    + (size_t)bb * sCb + (size_t)hh * sCh;
    _Float16* Ch = (_Float16*)Cout + (size_t)bb * sCb + (size_t)hh * sCh;
    const float* R = resid ? resid + (size_t)bb * sRb : nullptr;

    const int lane = threadIdx.x & 31;
    const int wave = threadIdx.x >> 5;
    const int m0 = (blockIdx.y * 8 + wave) * 32;
    const int n0 = nbase + blockIdx.x * (NT * 16);
    if (m0 >= M) return;                    // wave-uniform exit

    const int khalf = (lane & 16) ? 8 : 0;  // lane-group K offset

    int ra0 = m0 + (lane & 15);        if (ra0 >= M) ra0 = M - 1;
    int ra1 = m0 + 16 + (lane & 15);   if (ra1 >= M) ra1 = M - 1;
    const _Float16* pa0 = A + (size_t)ra0 * lda + khalf;
    const _Float16* pa1 = A + (size_t)ra1 * lda + khalf;

    const _Float16* pb[NT];
#pragma unroll
    for (int t = 0; t < NT; ++t) {
        int c = n0 + t * 16 + (lane & 15);
        if (c >= N) c = N - 1;
        pb[t] = Bt + (size_t)c * ldb + khalf;
    }

    const v8f vzero = {};
    v8f acc0[NT], acc1[NT];
#pragma unroll
    for (int t = 0; t < NT; ++t) { acc0[t] = vzero; acc1[t] = vzero; }

    struct FragSet { v16h a0, a1, b[NT]; };

    auto load_set = [&](FragSet& f, int koff) {
        f.a0 = ldfrag(pa0 + koff);
        f.a1 = ldfrag(pa1 + koff);
#pragma unroll
        for (int t = 0; t < NT; ++t) f.b[t] = ldfrag(pb[t] + koff);
    };
    auto compute = [&](const FragSet& f) {
#pragma unroll
        for (int t = 0; t < NT; ++t) {
            acc0[t] = wmma_f16(f.a0, f.b[t], acc0[t]);
            acc1[t] = wmma_f16(f.a1, f.b[t], acc1[t]);
        }
    };

    FragSet s0, s1;
    const int ns = K >> 5;               // number of 32-wide K steps (K % 32 == 0)
    load_set(s0, 0);
    int step = 0;
    // Ping-pong pairs: prefetch step+1 into the idle set, compute the live set.
    while (step + 2 <= ns - 1) {
        load_set(s1, 32);
        compute(s0);
        pa0 += 64; pa1 += 64;
#pragma unroll
        for (int t = 0; t < NT; ++t) pb[t] += 64;
        load_set(s0, 0);
        compute(s1);
        step += 2;
    }
    if (ns - step == 2) {
        load_set(s1, 32);
        compute(s0);
        compute(s1);
    } else {
        compute(s0);
    }

    // Epilogue.  C/D layout: lane -> col = (lane&15); VGPR r -> row = r + (lane>=16 ? 8:0)
    const int rb = m0 + ((lane & 16) ? 8 : 0);
#pragma unroll
    for (int t = 0; t < NT; ++t) {
        const int col = n0 + t * 16 + (lane & 15);
        if (col >= N) continue;
        const float bv = (flags & EPI_BIAS) ? bias[col] : 0.0f;
#pragma unroll
        for (int half = 0; half < 2; ++half) {
#pragma unroll
            for (int r = 0; r < 8; ++r) {
                const int row = rb + half * 16 + r;
                if (row >= M) continue;
                float v = half ? acc1[t][r] : acc0[t][r];
                if (flags & EPI_SCALE) v *= scale;
                v += bv;
                if (flags & EPI_GELU)
                    v = 0.5f * v * (1.0f + erff(v * 0.70710678118654752f));
                const size_t ci = (size_t)row * ldc + col;
                if (flags & EPI_RESID) v += R[ci];
                if (flags & EPI_OUT_F16) Ch[ci] = (_Float16)v;
                else                     Cf[ci] = v;
            }
        }
    }
}

// f32 [K,N] -> f16 [N,K] (transpose + convert), one thread per output element
__global__ void k_transpose_cvt(const float* __restrict__ in, _Float16* __restrict__ out,
                                int K, int N) {
    size_t i = (size_t)blockIdx.x * 256 + threadIdx.x;
    if (i >= (size_t)K * N) return;
    int n = (int)(i / K), k = (int)(i % K);
    out[i] = (_Float16)in[(size_t)k * N + n];
}

// plain f32 -> f16 convert
__global__ void k_cvt(const float* __restrict__ in, _Float16* __restrict__ out, size_t cnt) {
    size_t i = (size_t)blockIdx.x * 256 + threadIdx.x;
    if (i < cnt) out[i] = (_Float16)in[i];
}

// x [B,C,224,224] -> A_patch f16 [B*196, 768] (row = patch, col = (c*16+p)*16+q)
__global__ void k_patchify(const float* __restrict__ x, _Float16* __restrict__ A) {
    size_t i = (size_t)blockIdx.x * 256 + threadIdx.x;
    const size_t total = (size_t)BATCH * NPATCH * (CIN * PSZ * PSZ);
    if (i >= total) return;
    int col = (int)(i % (CIN * PSZ * PSZ));
    size_t rowi = i / (CIN * PSZ * PSZ);
    int wp = (int)(rowi % HPD);
    int hp = (int)((rowi / HPD) % HPD);
    int b  = (int)(rowi / (HPD * HPD));
    int c = col >> 8, p = (col >> 4) & 15, q = col & 15;
    size_t xi = (((size_t)b * CIN + c) * IMGS + (hp * PSZ + p)) * IMGS + (wp * PSZ + q);
    A[i] = (_Float16)x[xi];
}

// h[b,s,e] = token/patch + positions
__global__ void k_assemble(const float* __restrict__ embed, const float* __restrict__ cls,
                           const float* __restrict__ dist, const float* __restrict__ pos,
                           float* __restrict__ h) {
    size_t i = (size_t)blockIdx.x * 256 + threadIdx.x;
    const size_t total = (size_t)BATCH * SEQ * EMB;
    if (i >= total) return;
    int e = (int)(i % EMB);
    int s = (int)((i / EMB) % SEQ);
    int b = (int)(i / ((size_t)EMB * SEQ));
    float v;
    if (s == 0)      v = cls[e];
    else if (s == 1) v = dist[e];
    else             v = embed[((size_t)b * NPATCH + (s - 2)) * EMB + e];
    h[i] = v + pos[(size_t)s * EMB + e];
}

// LayerNorm: one block per row (E=768), out f16
__global__ __launch_bounds__(256)
void k_layernorm(const float* __restrict__ x, const float* __restrict__ g,
                 const float* __restrict__ bta, _Float16* __restrict__ out) {
    const int row = blockIdx.x;
    const float* px = x + (size_t)row * EMB;
    __shared__ float red[256];
    const int t = threadIdx.x;
    float s = 0.f;
    for (int e = t; e < EMB; e += 256) s += px[e];
    red[t] = s; __syncthreads();
    for (int st = 128; st > 0; st >>= 1) { if (t < st) red[t] += red[t + st]; __syncthreads(); }
    const float mean = red[0] / EMB;
    __syncthreads();
    float s2 = 0.f;
    for (int e = t; e < EMB; e += 256) { float d = px[e] - mean; s2 += d * d; }
    red[t] = s2; __syncthreads();
    for (int st = 128; st > 0; st >>= 1) { if (t < st) red[t] += red[t + st]; __syncthreads(); }
    const float rstd = rsqrtf(red[0] / EMB + LN_EPS);
    _Float16* po = out + (size_t)row * EMB;
    for (int e = t; e < EMB; e += 256)
        po[e] = (_Float16)((px[e] - mean) * rstd * g[e] + bta[e]);
}

// Softmax: one block per (b,h,q) row; reads f32 scores [.., SPAD], writes f16
// attn with zero padding for cols in [SEQ, SPAD).
__global__ __launch_bounds__(256)
void k_softmax(const float* __restrict__ scores, _Float16* __restrict__ attn) {
    const size_t row = blockIdx.x;
    const float* ps = scores + row * SPAD;
    _Float16* pa = attn + row * SPAD;
    const int t = threadIdx.x;
    __shared__ float red[256];
    float v = (t < SEQ) ? ps[t] : -3.0e38f;
    red[t] = v; __syncthreads();
    for (int st = 128; st > 0; st >>= 1) { if (t < st) red[t] = fmaxf(red[t], red[t + st]); __syncthreads(); }
    const float mx = red[0];
    __syncthreads();
    float ev = (t < SEQ) ? __expf(v - mx) : 0.0f;
    red[t] = ev; __syncthreads();
    for (int st = 128; st > 0; st >>= 1) { if (t < st) red[t] += red[t + st]; __syncthreads(); }
    const float inv = 1.0f / red[0];
    if (t < SPAD) pa[t] = (_Float16)(ev * inv);
}

// v f16 [B,S,E] -> vT f16 [B,NH,HD,SPAD] (zero-padded cols)
__global__ void k_transpose_v(const _Float16* __restrict__ v, _Float16* __restrict__ vT) {
    size_t i = (size_t)blockIdx.x * 256 + threadIdx.x;
    const size_t total = (size_t)BATCH * NHEAD * HDIM * SPAD;
    if (i >= total) return;
    int sp = (int)(i % SPAD);
    int d  = (int)((i / SPAD) % HDIM);
    int hh = (int)((i / ((size_t)SPAD * HDIM)) % NHEAD);
    int b  = (int)(i / ((size_t)SPAD * HDIM * NHEAD));
    vT[i] = (sp < SEQ) ? v[((size_t)b * SEQ + sp) * EMB + hh * HDIM + d] : (_Float16)0.0f;
}

// Final classifier heads: LN + [E,NC] matmul.  256 threads total (2*B*NC).
__global__ __launch_bounds__(64)
void k_heads(const float* __restrict__ h,
             const float* __restrict__ cg, const float* __restrict__ cbb,
             const float* __restrict__ cw, const float* __restrict__ cbs,
             const float* __restrict__ dg, const float* __restrict__ dbb,
             const float* __restrict__ dw, const float* __restrict__ dbs,
             float* __restrict__ out) {
    int i = blockIdx.x * 64 + threadIdx.x;
    if (i >= 2 * BATCH * NCLS) return;
    int n = i % NCLS;
    int b = (i / NCLS) % BATCH;
    int which = i / (NCLS * BATCH);            // 0 = cls (t[:,0]), 1 = dist (t[:,1])
    const float* row = h + ((size_t)b * SEQ + which) * EMB;
    float m = 0.f;
    for (int e = 0; e < EMB; ++e) m += row[e];
    m /= EMB;
    float var = 0.f;
    for (int e = 0; e < EMB; ++e) { float d = row[e] - m; var += d * d; }
    const float rstd = rsqrtf(var / EMB + LN_EPS);
    const float* g  = which ? dg  : cg;
    const float* bb = which ? dbb : cbb;
    const float* w  = which ? dw  : cw;
    float acc = which ? dbs[n] : cbs[n];
    for (int e = 0; e < EMB; ++e)
        acc += ((row[e] - m) * rstd * g[e] + bb[e]) * w[(size_t)e * NCLS + n];
    out[(size_t)which * BATCH * NCLS + (size_t)b * NCLS + n] = acc;
}

// ---------------------------------------------------------------------------
// Host side
// ---------------------------------------------------------------------------
struct GemmArgs {
    const _Float16* A; const _Float16* Bt; void* C;
    const float* resid; const float* bias;
    int M, N, K, lda, ldb, ldc;
    long long sAb, sAh, sBb, sBh, sCb, sCh, sRb;
    int Z, nhdiv; float scale; int flags;
};

template <int NT>
static inline void launch_gemm_nt(hipStream_t st, const GemmArgs& g, int nblk, int nbase) {
    dim3 grid(nblk, (g.M + 255) / 256, g.Z);
    k_gemm_wmma<NT><<<grid, 256, 0, st>>>(g.A, g.Bt, g.C, g.resid, g.bias,
                                          g.M, g.N, g.K, g.lda, g.ldb, g.ldc, nbase,
                                          g.sAb, g.sAh, g.sBb, g.sBh,
                                          g.sCb, g.sCh, g.sRb,
                                          g.nhdiv, g.scale, g.flags);
}

static inline void launch_gemm(hipStream_t st, const GemmArgs& g) {
    const int nfull = g.N / 64;
    if (nfull > 0) launch_gemm_nt<4>(st, g, nfull, 0);
    const int rem = g.N - nfull * 64;
    if (rem > 0) {
        const int nt = (rem + 15) / 16;
        switch (nt) {
            case 1: launch_gemm_nt<1>(st, g, 1, nfull * 64); break;
            case 2: launch_gemm_nt<2>(st, g, 1, nfull * 64); break;
            default: launch_gemm_nt<3>(st, g, 1, nfull * 64); break;
        }
    }
}

extern "C" void kernel_launch(void* const* d_in, const int* in_sizes, int n_in,
                              void* d_out, int out_size, void* d_ws, size_t ws_size,
                              hipStream_t stream) {
    (void)in_sizes; (void)n_in; (void)out_size; (void)ws_size;
    const float* x        = (const float*)d_in[0];
    const float* conv_w   = (const float*)d_in[1];
    const float* conv_b   = (const float*)d_in[2];
    const float* cls_tok  = (const float*)d_in[3];
    const float* dist_tok = (const float*)d_in[4];
    const float* pos      = (const float*)d_in[5];
    const float* ln1_g    = (const float*)d_in[6];
    const float* ln1_b    = (const float*)d_in[7];
    const float* wq = (const float*)d_in[8];  const float* bq = (const float*)d_in[9];
    const float* wk = (const float*)d_in[10]; const float* bk = (const float*)d_in[11];
    const float* wv = (const float*)d_in[12]; const float* bv = (const float*)d_in[13];
    const float* wo = (const float*)d_in[14]; const float* bo = (const float*)d_in[15];
    const float* ln2_g = (const float*)d_in[16];
    const float* ln2_b = (const float*)d_in[17];
    const float* w1 = (const float*)d_in[18]; const float* b1 = (const float*)d_in[19];
    const float* w2 = (const float*)d_in[20]; const float* b2 = (const float*)d_in[21];
    const float* cls_ln_g = (const float*)d_in[22];
    const float* cls_ln_b = (const float*)d_in[23];
    const float* cls_w    = (const float*)d_in[24];
    const float* cls_b    = (const float*)d_in[25];
    const float* dist_ln_g = (const float*)d_in[26];
    const float* dist_ln_b = (const float*)d_in[27];
    const float* dist_w    = (const float*)d_in[28];
    const float* dist_b    = (const float*)d_in[29];

    const int MROWS = BATCH * SEQ;        // 6336
    const int MPAT  = BATCH * NPATCH;     // 6272

    char* wsp = (char*)d_ws;
    auto alloc = [&](size_t bytes) -> void* {
        void* p = (void*)wsp;
        wsp += (bytes + 255) & ~(size_t)255;
        return p;
    };
    float*     hbuf   = (float*)alloc((size_t)MROWS * EMB * 4);
    _Float16*  zbuf   = (_Float16*)alloc((size_t)MROWS * EMB * 2);
    _Float16*  qbuf   = (_Float16*)alloc((size_t)MROWS * EMB * 2);
    _Float16*  kbuf   = (_Float16*)alloc((size_t)MROWS * EMB * 2);
    _Float16*  vbuf   = (_Float16*)alloc((size_t)MROWS * EMB * 2);
    _Float16*  obuf   = (_Float16*)alloc((size_t)MROWS * EMB * 2);
    _Float16*  vTbuf  = (_Float16*)alloc((size_t)BATCH * NHEAD * HDIM * SPAD * 2);
    float*     scores = (float*)alloc((size_t)BATCH * NHEAD * SEQ * SPAD * 4);
    _Float16*  attn   = (_Float16*)alloc((size_t)BATCH * NHEAD * SEQ * SPAD * 2);
    _Float16*  f1buf  = (_Float16*)alloc((size_t)MROWS * FFD * 2);
    _Float16*  patchA = (_Float16*)alloc((size_t)MPAT * (CIN * PSZ * PSZ) * 2);
    float*     embed  = (float*)alloc((size_t)MPAT * EMB * 4);
    _Float16*  convWh = (_Float16*)alloc((size_t)EMB * (CIN * PSZ * PSZ) * 2);
    _Float16*  wqT    = (_Float16*)alloc((size_t)EMB * EMB * 2);
    _Float16*  wkT    = (_Float16*)alloc((size_t)EMB * EMB * 2);
    _Float16*  wvT    = (_Float16*)alloc((size_t)EMB * EMB * 2);
    _Float16*  woT    = (_Float16*)alloc((size_t)EMB * EMB * 2);
    _Float16*  w1T    = (_Float16*)alloc((size_t)EMB * FFD * 2);
    _Float16*  w2T    = (_Float16*)alloc((size_t)FFD * EMB * 2);

    auto nb = [](size_t cnt) { return (unsigned)((cnt + 255) / 256); };

    // ---- Patch embedding: conv(stride==kernel) as GEMM ----
    const size_t convWn = (size_t)EMB * CIN * PSZ * PSZ;
    k_cvt<<<nb(convWn), 256, 0, stream>>>(conv_w, convWh, convWn);   // already [N=E, K=768]
    k_patchify<<<nb((size_t)MPAT * CIN * PSZ * PSZ), 256, 0, stream>>>(x, patchA);
    {
        GemmArgs g = { patchA, convWh, embed, nullptr, conv_b,
                       MPAT, EMB, CIN * PSZ * PSZ, CIN * PSZ * PSZ, CIN * PSZ * PSZ, EMB,
                       0,0,0,0,0,0,0, 1, 1, 1.0f, EPI_BIAS };
        launch_gemm(stream, g);
    }
    k_assemble<<<nb((size_t)MROWS * EMB), 256, 0, stream>>>(embed, cls_tok, dist_tok, pos, hbuf);

    const float attn_scale = 0.125f;   // 1/sqrt(64)
    const long long sE2 = (long long)EMB * EMB;

    for (int l = 0; l < DEPTHL; ++l) {
        // LN1
        k_layernorm<<<MROWS, 256, 0, stream>>>(hbuf, ln1_g + (size_t)l * EMB,
                                               ln1_b + (size_t)l * EMB, zbuf);
        // weights -> f16 transposed [N,K]
        k_transpose_cvt<<<nb(sE2), 256, 0, stream>>>(wq + (size_t)l * sE2, wqT, EMB, EMB);
        k_transpose_cvt<<<nb(sE2), 256, 0, stream>>>(wk + (size_t)l * sE2, wkT, EMB, EMB);
        k_transpose_cvt<<<nb(sE2), 256, 0, stream>>>(wv + (size_t)l * sE2, wvT, EMB, EMB);
        k_transpose_cvt<<<nb(sE2), 256, 0, stream>>>(wo + (size_t)l * sE2, woT, EMB, EMB);

        // q = LN(h) @ Wq + bq ; (reference quirk) k = q @ Wk + bk ; v = q @ Wv + bv
        {
            GemmArgs g = { zbuf, wqT, qbuf, nullptr, bq + (size_t)l * EMB,
                           MROWS, EMB, EMB, EMB, EMB, EMB,
                           0,0,0,0,0,0,0, 1, 1, 1.0f, EPI_BIAS | EPI_OUT_F16 };
            launch_gemm(stream, g);
        }
        {
            GemmArgs g = { qbuf, wkT, kbuf, nullptr, bk + (size_t)l * EMB,
                           MROWS, EMB, EMB, EMB, EMB, EMB,
                           0,0,0,0,0,0,0, 1, 1, 1.0f, EPI_BIAS | EPI_OUT_F16 };
            launch_gemm(stream, g);
        }
        {
            GemmArgs g = { qbuf, wvT, vbuf, nullptr, bv + (size_t)l * EMB,
                           MROWS, EMB, EMB, EMB, EMB, EMB,
                           0,0,0,0,0,0,0, 1, 1, 1.0f, EPI_BIAS | EPI_OUT_F16 };
            launch_gemm(stream, g);
        }

        k_transpose_v<<<nb((size_t)BATCH * NHEAD * HDIM * SPAD), 256, 0, stream>>>(vbuf, vTbuf);

        // scores[b,h] = (q_h @ k_h^T) * scale   (K = HDIM)
        {
            GemmArgs g = { qbuf, kbuf, scores, nullptr, nullptr,
                           SEQ, SEQ, HDIM, EMB, EMB, SPAD,
                           (long long)SEQ * EMB, HDIM,
                           (long long)SEQ * EMB, HDIM,
                           (long long)NHEAD * SEQ * SPAD, (long long)SEQ * SPAD, 0,
                           BATCH * NHEAD, NHEAD, attn_scale, EPI_SCALE };
            launch_gemm(stream, g);
        }

        k_softmax<<<BATCH * NHEAD * SEQ, 256, 0, stream>>>(scores, attn);

        // o[b,h] = attn @ V   (K = SPAD, zero-padded)
        {
            GemmArgs g = { attn, vTbuf, obuf, nullptr, nullptr,
                           SEQ, HDIM, SPAD, SPAD, SPAD, EMB,
                           (long long)NHEAD * SEQ * SPAD, (long long)SEQ * SPAD,
                           (long long)NHEAD * HDIM * SPAD, (long long)HDIM * SPAD,
                           (long long)SEQ * EMB, HDIM, 0,
                           BATCH * NHEAD, NHEAD, 1.0f, EPI_OUT_F16 };
            launch_gemm(stream, g);
        }

        // h = h + o @ Wo + bo  (residual, f32, in place)
        {
            GemmArgs g = { obuf, woT, hbuf, hbuf, bo + (size_t)l * EMB,
                           MROWS, EMB, EMB, EMB, EMB, EMB,
                           0,0,0,0,0,0,0, 1, 1, 1.0f, EPI_BIAS | EPI_RESID };
            launch_gemm(stream, g);
        }

        // LN2 + FFN
        k_layernorm<<<MROWS, 256, 0, stream>>>(hbuf, ln2_g + (size_t)l * EMB,
                                               ln2_b + (size_t)l * EMB, zbuf);
        k_transpose_cvt<<<nb((size_t)EMB * FFD), 256, 0, stream>>>(
            w1 + (size_t)l * EMB * FFD, w1T, EMB, FFD);
        {
            GemmArgs g = { zbuf, w1T, f1buf, nullptr, b1 + (size_t)l * FFD,
                           MROWS, FFD, EMB, EMB, EMB, FFD,
                           0,0,0,0,0,0,0, 1, 1, 1.0f, EPI_BIAS | EPI_GELU | EPI_OUT_F16 };
            launch_gemm(stream, g);
        }
        k_transpose_cvt<<<nb((size_t)FFD * EMB), 256, 0, stream>>>(
            w2 + (size_t)l * FFD * EMB, w2T, FFD, EMB);
        {
            GemmArgs g = { f1buf, w2T, hbuf, hbuf, b2 + (size_t)l * EMB,
                           MROWS, EMB, FFD, FFD, FFD, EMB,
                           0,0,0,0,0,0,0, 1, 1, 1.0f, EPI_BIAS | EPI_RESID };
            launch_gemm(stream, g);
        }
    }

    // Final heads: cls then dist, concatenated in d_out
    k_heads<<<(2 * BATCH * NCLS + 63) / 64, 64, 0, stream>>>(
        hbuf, cls_ln_g, cls_ln_b, cls_w, cls_b,
        dist_ln_g, dist_ln_b, dist_w, dist_b, (float*)d_out);
}
